// SelfAttn_30064771072667
// MI455X (gfx1250) — matrix operand order repaired and checked by reading, not compiled, not run on hardware
//
#include <hip/hip_runtime.h>
#include <math.h>

// Self-attention (B=4, C=256, N=64*64, d=32).
// Projections + energy: fp32 via V_WMMA_F32_16X16X4_F32 (full precision into
// softmax). Out-GEMM att*V^T (34.4 GF, dominant): split-bf16 via
// V_WMMA_F32_16X16X32_BF16: (ah+al)*(bh+bl) ~= ah*bh + ah*bl + al*bh,
// fp32 accumulate -> ~1e-4 rel error at 2.7x fewer matrix instructions.

typedef __attribute__((ext_vector_type(2)))  float  v2f;
typedef __attribute__((ext_vector_type(8)))  float  v8f;
typedef __attribute__((ext_vector_type(16))) __bf16 v16bf;

#define B_   4
#define C_   256
#define QKD  32
#define N_   4096          // 64*64 spatial positions
#define EP   4100          // LDS row pitch (words): 4100%64==4 -> conflict-free

__device__ __forceinline__ v8f wmma_f32(v2f a, v2f b, v8f c) {
    return __builtin_amdgcn_wmma_f32_16x16x4_f32(false, a, false, b,
                                                 (short)0, c, false, false);
}
__device__ __forceinline__ v8f wmma_bf16(v16bf a, v16bf b, v8f c) {
    return __builtin_amdgcn_wmma_f32_16x16x32_bf16(false, a, false, b,
                                                   (short)0, c, false, false);
}

// round-to-nearest-even f32 -> bf16 bits
__device__ __forceinline__ unsigned int f32_to_bf16(float f) {
    unsigned int u = __float_as_uint(f);
    return (u + 0x7fffu + ((u >> 16) & 1u)) >> 16;
}
__device__ __forceinline__ float bf16_to_f32(unsigned int h) {
    return __uint_as_float(h << 16);
}

// Load bf16 [p, p+8) and [p+16, p+24) (16B-aligned each) as a v16bf fragment.
__device__ __forceinline__ v16bf load_v16bf(const unsigned short* p) {
    uint4 a = *(const uint4*)p;
    uint4 b = *(const uint4*)(p + 16);
    union { unsigned int u[8]; v16bf v; } t;
    t.u[0] = a.x; t.u[1] = a.y; t.u[2] = a.z; t.u[3] = a.w;
    t.u[4] = b.x; t.u[5] = b.y; t.u[6] = b.z; t.u[7] = b.w;
    return t.v;
}

// ---------------------------------------------------------------------------
// Fused 1x1-conv projection: out[n,o] = relu(sum_c W[o,c]*x[b,c,n] + bias[o])
// One 16(n) x 16(o) tile per wave (blockDim=256 -> 8 waves).
// mode 0: fp32 (N, OD) row-major   (Q)
// mode 1: fp32 (OD, N)             (K)
// mode 2: bf16 hi/lo planes (OD,N) (V, feeds split-bf16 out-GEMM)
// ---------------------------------------------------------------------------
__global__ void proj_kernel(const float* __restrict__ x,
                            const float* __restrict__ Wmat,
                            const float* __restrict__ bias,
                            float* __restrict__ outbuf,
                            unsigned short* __restrict__ out_h,
                            unsigned short* __restrict__ out_l,
                            int OD, int mode)
{
    const int wave = threadIdx.x >> 5;
    const int lane = threadIdx.x & 31;
    const int hl   = lane >> 4;
    const int l16  = lane & 15;

    const int tiles_per_batch = (N_ / 16) * (OD / 16);
    const int tile = blockIdx.x * 8 + wave;
    const int b    = tile / tiles_per_batch;
    const int t    = tile % tiles_per_batch;
    const int m0   = (t % (N_ / 16)) * 16;
    const int o0   = (t / (N_ / 16)) * 16;

    const float* __restrict__ xb = x + (size_t)b * C_ * N_;

    v8f acc = {};
    for (int c = 0; c < C_; c += 4) {
        v2f a, bf;
        a.x  = xb[(size_t)(c + 2 * hl + 0) * N_ + m0 + l16];
        a.y  = xb[(size_t)(c + 2 * hl + 1) * N_ + m0 + l16];
        bf.x = Wmat[(size_t)(o0 + l16) * C_ + c + 2 * hl + 0];
        bf.y = Wmat[(size_t)(o0 + l16) * C_ + c + 2 * hl + 1];
        acc = wmma_f32(a, bf, acc);
    }

    const float bb = bias[o0 + l16];
    #pragma unroll
    for (int r = 0; r < 8; ++r) {
        const int m = m0 + r + 8 * hl;
        float v = acc[r] + bb;
        v = v > 0.0f ? v : 0.0f;                 // ReLU
        if (mode == 2) {
            const size_t idx = (size_t)b * N_ * OD + (size_t)(o0 + l16) * N_ + m;
            const unsigned int hi = f32_to_bf16(v);
            const float rf = v - bf16_to_f32(hi);
            out_h[idx] = (unsigned short)hi;
            out_l[idx] = (unsigned short)f32_to_bf16(rf);
        } else if (mode == 1) {
            outbuf[(size_t)b * N_ * OD + (size_t)(o0 + l16) * N_ + m] = v;
        } else {
            outbuf[(size_t)b * N_ * OD + (size_t)m * OD + o0 + l16]   = v;
        }
    }
}

// ---------------------------------------------------------------------------
// Per workgroup: one (batch, 16-row) tile of attention. 512 thr = 16 waves.
//   Phase 1: E[16][4096] = Q_tile * K (fp32 WMMA) into 256KB LDS
//   Phase 2: row softmax (1 wave/row); NT-store fp32 att; repack each LDS
//            element in place as (bf16_hi | bf16_lo<<16)
//   Phase 3: out = att * V^T with split-bf16 WMMA (A from LDS, B = bf16
//            hi/lo planes, L2-resident); out = gamma*acc + x
// ---------------------------------------------------------------------------
__global__ void attn_kernel(const float* __restrict__ Q,            // (B,N,32)
                            const float* __restrict__ K,            // (B,32,N)
                            const unsigned short* __restrict__ Vh,  // (B,C,N)
                            const unsigned short* __restrict__ Vl,  // (B,C,N)
                            const float* __restrict__ x,            // (B,C,N)
                            const float* __restrict__ gamma_p,
                            float* __restrict__ out,                // (B,C,N)
                            float* __restrict__ att)                // (B,N,N)
{
    extern __shared__ float e[];                 // [16][EP]
    unsigned int* eu = (unsigned int*)e;
    const int wave = threadIdx.x >> 5;           // 0..15
    const int lane = threadIdx.x & 31;
    const int hl   = lane >> 4;
    const int l16  = lane & 15;

    const int b  = blockIdx.x / (N_ / 16);
    const int m0 = (blockIdx.x % (N_ / 16)) * 16;

    const float* __restrict__ Qb = Q + (size_t)b * N_ * QKD;
    const float* __restrict__ Kb = K + (size_t)b * QKD * N_;

    // 8 A-fragments of the Q row-tile (16x32), reused for all 256 col tiles.
    v2f aq[8];
    #pragma unroll
    for (int kk = 0; kk < 8; ++kk) {
        aq[kk].x = Qb[(size_t)(m0 + l16) * QKD + kk * 4 + 2 * hl + 0];
        aq[kk].y = Qb[(size_t)(m0 + l16) * QKD + kk * 4 + 2 * hl + 1];
    }

    // ---- Phase 1: energy row-block into LDS (fp32 WMMA) --------------------
    for (int nt = wave; nt < N_ / 16; nt += 16) {
        const int n0 = nt * 16;
        v8f acc = {};
        #pragma unroll
        for (int kk = 0; kk < 8; ++kk) {
            v2f bf;
            bf.x = Kb[(size_t)(kk * 4 + 2 * hl + 0) * N_ + n0 + l16];
            bf.y = Kb[(size_t)(kk * 4 + 2 * hl + 1) * N_ + n0 + l16];
            acc = wmma_f32(aq[kk], bf, acc);
        }
        #pragma unroll
        for (int r = 0; r < 8; ++r)
            e[(r + 8 * hl) * EP + n0 + l16] = acc[r];
    }
    __syncthreads();

    // ---- Phase 2: softmax, one wave per row; in-place bf16 hi/lo repack ----
    {
        const int row = wave;
        float mx = -INFINITY;
        for (int col = lane; col < N_; col += 32)
            mx = fmaxf(mx, e[row * EP + col]);
        #pragma unroll
        for (int off = 16; off; off >>= 1)
            mx = fmaxf(mx, __shfl_xor(mx, off, 32));

        float s = 0.0f;
        for (int col = lane; col < N_; col += 32) {
            const float ev = __expf(e[row * EP + col] - mx);
            e[row * EP + col] = ev;
            s += ev;
        }
        #pragma unroll
        for (int off = 16; off; off >>= 1)
            s += __shfl_xor(s, off, 32);

        const float inv = 1.0f / s;
        float* __restrict__ arow =
            att + (size_t)b * N_ * N_ + (size_t)(m0 + row) * N_;
        for (int col = lane; col < N_; col += 32) {
            const float av = e[row * EP + col] * inv;
            __builtin_nontemporal_store(av, &arow[col]);  // 268MB stream
            const unsigned int hi = f32_to_bf16(av);
            const float rf = av - bf16_to_f32(hi);
            const unsigned int lo = f32_to_bf16(rf);
            eu[row * EP + col] = hi | (lo << 16);   // same-address overwrite
        }
    }
    __syncthreads();

    // ---- Phase 3: out[m,c] = sum_n att[m,n]*V[c,n], split-bf16 WMMA --------
    const int c0 = wave * 16;                    // each wave owns 16 channels
    const unsigned short* __restrict__ Vhb = Vh + (size_t)b * C_ * N_;
    const unsigned short* __restrict__ Vlb = Vl + (size_t)b * C_ * N_;
    const size_t vrow = (size_t)(c0 + l16) * N_;

    v8f acc = {};
    for (int n = 0; n < N_; n += 32) {
        // A (16x32 bf16): M = l16; VGPR j<4 -> K=2j+8*hl, j>=4 -> K=16+2(j-4)+8*hl.
        // LDS words hold (hi | lo<<16) of one att element each.
        union { unsigned int u[8]; v16bf v; } H, L;
        #pragma unroll
        for (int g = 0; g < 2; ++g) {
            const unsigned int* p = eu + l16 * EP + n + 16 * g + 8 * hl;
            uint4 q0 = *(const uint4*)p;
            uint4 q1 = *(const uint4*)(p + 4);
            unsigned int w[8] = {q0.x, q0.y, q0.z, q0.w, q1.x, q1.y, q1.z, q1.w};
            #pragma unroll
            for (int jj = 0; jj < 4; ++jj) {
                const unsigned int w0 = w[2 * jj], w1 = w[2 * jj + 1];
                H.u[4 * g + jj] = (w0 & 0xffffu) | (w1 << 16);
                L.u[4 * g + jj] = (w0 >> 16)     | (w1 & 0xffff0000u);
            }
        }
        // B (32x16 bf16): same K order as A: elems 0-7 -> K=8*hl+s,
        // elems 8-15 -> K=16+8*hl+(s-8); two 16B (C,N)-plane reads.
        const v16bf bh = load_v16bf(Vhb + vrow + n + 8 * hl);
        const v16bf bl = load_v16bf(Vlb + vrow + n + 8 * hl);
        acc = wmma_bf16(H.v, bh, acc);
        acc = wmma_bf16(H.v, bl, acc);
        acc = wmma_bf16(L.v, bh, acc);
    }

    const float g = gamma_p[0];
    const float* __restrict__ xb = x   + (size_t)b * C_ * N_;
    float*       __restrict__ ob = out + (size_t)b * C_ * N_;
    #pragma unroll
    for (int r = 0; r < 8; ++r) {
        const int m = m0 + r + 8 * hl;
        const size_t idx = (size_t)(c0 + l16) * N_ + m;
        ob[idx] = g * acc[r] + xb[idx];
    }
}

// ---------------------------------------------------------------------------
extern "C" void kernel_launch(void* const* d_in, const int* in_sizes, int n_in,
                              void* d_out, int out_size, void* d_ws, size_t ws_size,
                              hipStream_t stream)
{
    const float* x     = (const float*)d_in[0];
    const float* Wq    = (const float*)d_in[1];
    const float* bq    = (const float*)d_in[2];
    const float* Wk    = (const float*)d_in[3];
    const float* bk    = (const float*)d_in[4];
    const float* Wv    = (const float*)d_in[5];
    const float* bv    = (const float*)d_in[6];
    const float* gamma = (const float*)d_in[7];

    float* out = (float*)d_out;                        // (B, C, N)
    float* att = out + (size_t)B_ * C_ * N_;           // (B, N, N)

    float* qbuf = (float*)d_ws;                        // (B, N, 32) fp32  2MB
    float* kbuf = qbuf + (size_t)B_ * N_ * QKD;        // (B, 32, N) fp32  2MB
    unsigned short* vh = (unsigned short*)(kbuf + (size_t)B_ * QKD * N_); // 8MB
    unsigned short* vl = vh + (size_t)B_ * C_ * N_;                       // 8MB

    const int qk_blocks = (B_ * (N_ / 16) * (QKD / 16)) / 8;  // 256
    const int v_blocks  = (B_ * (N_ / 16) * (C_  / 16)) / 8;  // 2048

    proj_kernel<<<qk_blocks, 256, 0, stream>>>(x, Wq, bq, qbuf, nullptr, nullptr, QKD, 0);
    proj_kernel<<<qk_blocks, 256, 0, stream>>>(x, Wk, bk, kbuf, nullptr, nullptr, QKD, 1);
    proj_kernel<<<v_blocks,  256, 0, stream>>>(x, Wv, bv, nullptr, vh, vl, C_, 2);

    attn_kernel<<<B_ * (N_ / 16), 512, 16 * EP * sizeof(float), stream>>>(
        qbuf, kbuf, vh, vl, x, gamma, out, att);
}
